// DCNv3Bottleneck_58935541236115
// MI455X (gfx1250) — matrix operand order repaired and checked by reading, not compiled, not run on hardware
//
#include <hip/hip_runtime.h>
#include <hip/hip_bf16.h>

typedef _Float16 h16;
typedef __attribute__((ext_vector_type(16))) _Float16 v16h;
typedef __attribute__((ext_vector_type(8)))  _Float16 v8h;
typedef __attribute__((ext_vector_type(8)))  float    v8f;
typedef __attribute__((ext_vector_type(4)))  float    v4f;

#define NB    8
#define C1C   256
#define C2C   256
#define CMID  128
#define HH    56
#define WW    56
#define HWC   (HH*WW)          /* 3136 */
#define MROWS (NB*HWC)         /* 25088 */
#define HP    58
#define WP    58
#define GRP   4
#define EPSF  1e-5f

__device__ __forceinline__ float silu_f(float v) { return v / (1.f + __expf(-v)); }

// ---------------------------------------------------------------------------
// CDNA5 async global->LDS copy (ASYNCcnt-tracked, CDNA5 ISA).
// VDST = per-lane LDS byte address (low 32 bits of generic LDS pointer),
// VADDR = 64-bit global address, 16 bytes per lane.
// ---------------------------------------------------------------------------
__device__ __forceinline__ void async_b128(h16* dst_lds, const h16* src)
{
    unsigned           l = (unsigned)(uintptr_t)dst_lds;
    unsigned long long g = (unsigned long long)(uintptr_t)src;
    asm volatile("global_load_async_to_lds_b128 %0, %1, off" :: "v"(l), "v"(g) : "memory");
}
__device__ __forceinline__ void wait_async()
{
    asm volatile("s_wait_asynccnt 0" ::: "memory");
}

// ---------------------------------------------------------------------------
// Weight repack: f32 -> f16 GEMM B matrices, stored TRANSPOSED (N x K row-major)
// so that GEMM LDS staging is fully contiguous b128 traffic.
//   Bt1: 128x2304 (im2col conv1, k = (ky*3+kx)*256 + c)
//   Bt2: 128x128  (w_inp)
//   Bt3: 128x128  (w_off | w_msk | zeros)
//   Bt4: 256x128  (w_out)
//   bias_om: f32[128] (b_off | b_msk | 0)
// ---------------------------------------------------------------------------
__global__ __launch_bounds__(256)
void prep_weights(const float* __restrict__ w_cv1, const float* __restrict__ w_inp,
                  const float* __restrict__ w_off, const float* __restrict__ w_msk,
                  const float* __restrict__ w_out, const float* __restrict__ b_off,
                  const float* __restrict__ b_msk,
                  h16* __restrict__ Bt1, h16* __restrict__ Bt2, h16* __restrict__ Bt3,
                  h16* __restrict__ Bt4, float* __restrict__ bias_om)
{
    int i = blockIdx.x * 256 + threadIdx.x;
    const int n1 = 128 * 2304, n2 = 128 * 128, n3 = 128 * 128, n4 = 256 * 128;
    if (i < n1) {
        int o = i / 2304, k = i - o * 2304;
        int kidx = k >> 8, c = k & 255;
        int ky = kidx / 3, kx = kidx - ky * 3;
        Bt1[i] = (h16)w_cv1[((o * 256 + c) * 3 + ky) * 3 + kx];
        return;
    }
    i -= n1;
    if (i < n2) { int n = i >> 7, k = i & 127; Bt2[i] = (h16)w_inp[k * 128 + n]; return; }
    i -= n2;
    if (i < n3) {
        int n = i >> 7, k = i & 127;
        float v = 0.f;
        if (n < 72)        v = w_off[k * 72 + n];
        else if (n < 108)  v = w_msk[k * 36 + (n - 72)];
        Bt3[i] = (h16)v;
        return;
    }
    i -= n3;
    if (i < n4) { int n = i >> 7, k = i & 127; Bt4[i] = (h16)w_out[k * 256 + n]; return; }
    i -= n4;
    if (i < 128) {
        float v = 0.f;
        if (i < 72)       v = b_off[i];
        else if (i < 108) v = b_msk[i - 72];
        bias_om[i] = v;
    }
}

// NCHW f32 -> spatially padded NHWC f16 (8, 58, 58, 256); border pre-zeroed by
// memset so the conv im2col A-tiles need no bounds checks (pure async copies).
__global__ __launch_bounds__(256)
void x_to_nhwc(const float* __restrict__ x, h16* __restrict__ xh)
{
    size_t i = (size_t)blockIdx.x * 256 + threadIdx.x;   // over N*HW*C1 (nhwc order)
    int c = (int)(i & 255);
    size_t s = i >> 8;                                    // n*HW + y*W + x
    int n = (int)(s / HWC); int rem = (int)(s - (size_t)n * HWC);
    int y = rem / WW, xx = rem - (rem / WW) * WW;
    xh[(((size_t)n * HP + (y + 1)) * WP + (xx + 1)) * C1C + c] =
        (h16)x[((size_t)n * C1C + c) * HWC + rem];
}

// ---------------------------------------------------------------------------
// LDS fragment loaders following the ISA 16-bit WMMA VGPR layouts (wave32).
// LDS rows are 64 halves wide (BK=64); kk selects the 32-wide k-step.
// ---------------------------------------------------------------------------
__device__ __forceinline__ v16h ldA(const h16* As, int row16, int kk, int lane)
{
    const int kb = (lane >= 16) ? 8 : 0;
    const h16* ap = As + (row16 + (lane & 15)) * 64 + kk + kb;
    v8h alo = *(const v8h*)ap;
    v8h ahi = *(const v8h*)(ap + 16);
    v16h a;
#pragma unroll
    for (int i = 0; i < 8; i++) { a[i] = alo[i]; a[8 + i] = ahi[i]; }
    return a;
}

__device__ __forceinline__ v16h ldB(const h16* Bs, int col16, int kk, int lane)
{
    const int kob = (lane >= 16) ? 8 : 0;
    const h16* bp = Bs + (col16 + (lane & 15)) * 64 + kk + kob;
    v8h blo = *(const v8h*)bp;
    v8h bhi = *(const v8h*)(bp + 16);
    v16h b;
#pragma unroll
    for (int i = 0; i < 8; i++) { b[i] = blo[i]; b[8 + i] = bhi[i]; }
    return b;
}

// ---------------------------------------------------------------------------
// Tiled WMMA GEMM: BM=64, BN=128, BK=64; 256 threads = 8 waves.
// Wave (wm in 0..1, wn in 0..3) owns a 32x32 C tile -> 4 accumulators,
// 8 x v_wmma_f32_16x16x32_f16 per staging round.  f16 sources (conv-A, all B)
// are staged with async global->LDS b128 copies.
// MODE 0: A = im2col(padded x_nhwc f16) K=2304, epi BN1+SiLU -> h (MxCMID f32)
// MODE 1: A = h  f32, K=128, epilogue +b_inp                -> padded value
// MODE 2: A = t  f32, K=128, epilogue +bias_om              -> offmsk raw
// MODE 3: A = samp f32, K=128, epilogue +b_out,BN2,SiLU,+x  -> d_out NCHW
// ---------------------------------------------------------------------------
template <int MODE>
__global__ __launch_bounds__(256)
void gemm_wmma(const float* __restrict__ A, const h16* __restrict__ xh,
               const h16* __restrict__ Bt, int Kdim,
               const float* __restrict__ bias,
               const float* __restrict__ bng, const float* __restrict__ bnb,
               const float* __restrict__ bnm, const float* __restrict__ bnv,
               const float* __restrict__ xres, float* __restrict__ out)
{
    __shared__ h16 As[64 * 64];      // [row 64][k 64]
    __shared__ h16 Bs[128 * 64];     // transposed: [col 128][k 64]

    const int tid  = threadIdx.x;
    const int lane = tid & 31;
    const int wv   = tid >> 5;
    const int wm   = wv & 1;         // M half (32 rows)
    const int wn   = wv >> 1;        // N quarter (32 cols)
    const int m0   = blockIdx.y * 64;
    const int n0   = blockIdx.x * 128;

    const int ar = tid >> 2, aseg = (tid & 3) * 16;       // A staging: row, 16-half seg
    const int bcol = tid & 127, bhalf = (tid >> 7) * 32;  // B staging: col, 32-half seg

    // decode output pixel for conv im2col (row fixed per staging thread)
    int an = 0, ay = 0, ax = 0;
    {
        int row = m0 + ar;
        an = row / HWC; int rem = row - an * HWC; ay = rem / WW; ax = rem - ay * WW;
    }

    v8f acc00, acc01, acc10, acc11;
#pragma unroll
    for (int i = 0; i < 8; i++) { acc00[i] = 0.f; acc01[i] = 0.f; acc10[i] = 0.f; acc11[i] = 0.f; }

    for (int k0 = 0; k0 < Kdim; k0 += 64) {
        __syncthreads();
        // ---- stage A ----
        if (MODE == 0) {
            // padded im2col: within a 64-wide K tile, (ky,kx) is constant
            int kidx = k0 >> 8, c0 = k0 & 255;
            int ky = kidx / 3, kx = kidx - ky * 3;
            const h16* src = xh + ((((size_t)an * HP + (ay + ky)) * WP + (ax + kx)) * C1C) + c0 + aseg;
            h16* dst = As + ar * 64 + aseg;
            async_b128(dst,     src);
            async_b128(dst + 8, src + 8);
        } else {
            const float* ap = A + (size_t)(m0 + ar) * Kdim + k0 + aseg;
            v4f f0 = *(const v4f*)ap;
            v4f f1 = *(const v4f*)(ap + 4);
            v4f f2 = *(const v4f*)(ap + 8);
            v4f f3 = *(const v4f*)(ap + 12);
            v8h a0, a1;
#pragma unroll
            for (int i = 0; i < 4; i++) {
                a0[i] = (h16)f0[i]; a0[4 + i] = (h16)f1[i];
                a1[i] = (h16)f2[i]; a1[4 + i] = (h16)f3[i];
            }
            *(v8h*)(As + ar * 64 + aseg)     = a0;
            *(v8h*)(As + ar * 64 + aseg + 8) = a1;
        }
        // ---- stage B: 32 halves per thread, contiguous (Bt is N x K) ----
        {
            const h16* src = Bt + (size_t)(n0 + bcol) * Kdim + k0 + bhalf;
            h16* dst = Bs + bcol * 64 + bhalf;
            async_b128(dst,      src);
            async_b128(dst + 8,  src + 8);
            async_b128(dst + 16, src + 16);
            async_b128(dst + 24, src + 24);
        }
        wait_async();
        __syncthreads();

#pragma unroll
        for (int kk = 0; kk < 64; kk += 32) {
            v16h a0 = ldA(As, wm * 32,      kk, lane);
            v16h a1 = ldA(As, wm * 32 + 16, kk, lane);
            v16h b0 = ldB(Bs, wn * 32,      kk, lane);
            v16h b1 = ldB(Bs, wn * 32 + 16, kk, lane);
            acc00 = __builtin_amdgcn_wmma_f32_16x16x32_f16(false, a0, false, b0, (short)0, acc00, false, false);
            acc01 = __builtin_amdgcn_wmma_f32_16x16x32_f16(false, a0, false, b1, (short)0, acc01, false, false);
            acc10 = __builtin_amdgcn_wmma_f32_16x16x32_f16(false, a1, false, b0, (short)0, acc10, false, false);
            acc11 = __builtin_amdgcn_wmma_f32_16x16x32_f16(false, a1, false, b1, (short)0, acc11, false, false);
        }
    }

    // epilogue (C layout: vgpr r, lane l -> M = r + 8*(l>=16), N = l&15)
    const int mo = (lane >= 16) ? 8 : 0;
    const int nn = lane & 15;
#pragma unroll
    for (int ma = 0; ma < 2; ma++) {
#pragma unroll
        for (int nb = 0; nb < 2; nb++) {
#pragma unroll
            for (int r = 0; r < 8; r++) {
                float vacc;
                if (ma == 0) vacc = (nb == 0) ? acc00[r] : acc01[r];
                else         vacc = (nb == 0) ? acc10[r] : acc11[r];
                int grow = m0 + wm * 32 + ma * 16 + r + mo;
                int gcol = n0 + wn * 32 + nb * 16 + nn;
                if (MODE == 0) {
                    float sc = bng[gcol] * __frsqrt_rn(bnv[gcol] + EPSF);
                    float sh = bnb[gcol] - bnm[gcol] * sc;
                    out[(size_t)grow * CMID + gcol] = silu_f(vacc * sc + sh);
                } else if (MODE == 1) {
                    float t = vacc + bias[gcol];
                    int n = grow / HWC; int rem = grow - n * HWC;
                    int y = rem / WW, x = rem - (rem / WW) * WW;
                    out[(((size_t)n * HP + (y + 1)) * WP + (x + 1)) * CMID + gcol] = t;
                } else if (MODE == 2) {
                    out[(size_t)grow * 128 + gcol] = vacc + bias[gcol];
                } else {
                    float t = vacc + bias[gcol];
                    float sc = bng[gcol] * __frsqrt_rn(bnv[gcol] + EPSF);
                    float sh = bnb[gcol] - bnm[gcol] * sc;
                    t = silu_f(t * sc + sh);
                    int n = grow / HWC; int rem = grow - n * HWC;
                    int y = rem / WW, x = rem - (rem / WW) * WW;
                    size_t idx = (((size_t)n * C2C + gcol) * HH + y) * WW + x;
                    out[idx] = t + xres[idx];
                }
            }
        }
    }
}

// ---------------------------------------------------------------------------
// Depthwise 3x3 + bias -> LayerNorm(channel) -> exact GELU.  Wave per pixel,
// lane handles 4 channels; cross-lane reduction via shfl_xor (wave32).
// ---------------------------------------------------------------------------
__global__ __launch_bounds__(256)
void dw_ln_gelu(const float* __restrict__ hbuf, const float* __restrict__ wdw,
                const float* __restrict__ bdw, const float* __restrict__ lng,
                const float* __restrict__ lnb, float* __restrict__ tbuf)
{
    int wv = threadIdx.x >> 5, lane = threadIdx.x & 31;
    int p = blockIdx.x * 8 + wv;
    int n = p / HWC; int rem = p - n * HWC; int y = rem / WW; int x = rem - y * WW;
    float val[4];
#pragma unroll
    for (int q = 0; q < 4; q++) {
        int c = lane + 32 * q;
        float s = bdw[c];
#pragma unroll
        for (int ky = 0; ky < 3; ky++) {
            int yy = y + ky - 1; if (yy < 0 || yy >= HH) continue;
#pragma unroll
            for (int kx = 0; kx < 3; kx++) {
                int xx = x + kx - 1; if (xx < 0 || xx >= WW) continue;
                s += hbuf[(((size_t)n * HH + yy) * WW + xx) * CMID + c] * wdw[c * 9 + ky * 3 + kx];
            }
        }
        val[q] = s;
    }
    float sum = 0.f, sq = 0.f;
#pragma unroll
    for (int q = 0; q < 4; q++) { sum += val[q]; sq += val[q] * val[q]; }
    for (int off = 16; off; off >>= 1) {
        sum += __shfl_xor(sum, off, 32);
        sq  += __shfl_xor(sq,  off, 32);
    }
    float mean = sum * (1.f / 128.f);
    float var  = sq * (1.f / 128.f) - mean * mean;
    float inv  = __frsqrt_rn(var + EPSF);
#pragma unroll
    for (int q = 0; q < 4; q++) {
        int c = lane + 32 * q;
        float v = (val[q] - mean) * inv * lng[c] + lnb[c];
        v = 0.5f * v * (1.f + erff(v * 0.70710678118654752f));
        tbuf[(size_t)p * CMID + c] = v;
    }
}

// ---------------------------------------------------------------------------
// DCNv3 core: wave per (pixel, group); lane = channel in group.
// ---------------------------------------------------------------------------
__device__ __forceinline__ float vfetch(const float* __restrict__ v, int n, int g,
                                        int yi, int xi, int c)
{
    if (yi < 0 || yi >= HP || xi < 0 || xi >= WP) return 0.f;
    return v[(((size_t)n * HP + yi) * WP + xi) * CMID + g * 32 + c];
}

__global__ __launch_bounds__(256)
void dcn_sample(const float* __restrict__ om, const float* __restrict__ vbuf,
                float* __restrict__ samp)
{
    int wv = threadIdx.x >> 5, lane = threadIdx.x & 31;
    int item = blockIdx.x * 8 + wv;
    int g = item & 3, p = item >> 2;
    int n = p / HWC; int rem = p - n * HWC; int y = rem / WW; int x = rem - y * WW;

    const float* mrow = om + (size_t)p * 128 + 72 + g * 9;
    float ml[9], mx = -1e30f;
#pragma unroll
    for (int j = 0; j < 9; j++) { ml[j] = mrow[j]; mx = fmaxf(mx, ml[j]); }
    float den = 0.f;
#pragma unroll
    for (int j = 0; j < 9; j++) { ml[j] = __expf(ml[j] - mx); den += ml[j]; }
    float rden = 1.f / den;

    const float* orow = om + (size_t)p * 128 + g * 18;
    float acc = 0.f;
#pragma unroll
    for (int j = 0; j < 9; j++) {
        float sx = (float)(x + j / 3) + orow[2 * j];
        float sy = (float)(y + j % 3) + orow[2 * j + 1];
        float fx0 = floorf(sx), fy0 = floorf(sy);
        float wx = sx - fx0, wy = sy - fy0;
        int x0 = (int)fx0, y0 = (int)fy0;
        float v00 = vfetch(vbuf, n, g, y0,     x0,     lane);
        float v01 = vfetch(vbuf, n, g, y0,     x0 + 1, lane);
        float v10 = vfetch(vbuf, n, g, y0 + 1, x0,     lane);
        float v11 = vfetch(vbuf, n, g, y0 + 1, x0 + 1, lane);
        float bi = (1.f - wy) * ((1.f - wx) * v00 + wx * v01)
                 +        wy  * ((1.f - wx) * v10 + wx * v11);
        acc += (ml[j] * rden) * bi;
    }
    samp[(size_t)p * CMID + g * 32 + lane] = acc;
}

// ---------------------------------------------------------------------------
static inline size_t al256(size_t v) { return (v + 255) & ~(size_t)255; }

extern "C" void kernel_launch(void* const* d_in, const int* in_sizes, int n_in,
                              void* d_out, int out_size, void* d_ws, size_t ws_size,
                              hipStream_t stream)
{
    const float* x     = (const float*)d_in[0];
    const float* w_cv1 = (const float*)d_in[1];
    const float* bn1_g = (const float*)d_in[2];
    const float* bn1_b = (const float*)d_in[3];
    const float* bn1_m = (const float*)d_in[4];
    const float* bn1_v = (const float*)d_in[5];
    const float* w_dw  = (const float*)d_in[6];
    const float* b_dw  = (const float*)d_in[7];
    const float* ln_g  = (const float*)d_in[8];
    const float* ln_b  = (const float*)d_in[9];
    const float* w_inp = (const float*)d_in[10];
    const float* b_inp = (const float*)d_in[11];
    const float* w_off = (const float*)d_in[12];
    const float* b_off = (const float*)d_in[13];
    const float* w_msk = (const float*)d_in[14];
    const float* b_msk = (const float*)d_in[15];
    const float* w_out = (const float*)d_in[16];
    const float* b_out = (const float*)d_in[17];
    const float* bn2_g = (const float*)d_in[18];
    const float* bn2_b = (const float*)d_in[19];
    const float* bn2_m = (const float*)d_in[20];
    const float* bn2_v = (const float*)d_in[21];
    float* out = (float*)d_out;

    // workspace layout
    char* ws = (char*)d_ws;
    size_t off = 0;
    size_t xhbytes = (size_t)NB * HP * WP * C1C * 2;     // padded NHWC f16
    h16*   xh      = (h16*)(ws + off);  off = al256(off + xhbytes);
    h16*   Bt1     = (h16*)(ws + off);  off = al256(off + (size_t)128 * 2304 * 2);
    h16*   Bt2     = (h16*)(ws + off);  off = al256(off + (size_t)128 * 128 * 2);
    h16*   Bt3     = (h16*)(ws + off);  off = al256(off + (size_t)128 * 128 * 2);
    h16*   Bt4     = (h16*)(ws + off);  off = al256(off + (size_t)256 * 128 * 2);
    float* bias_om = (float*)(ws + off); off = al256(off + 128 * 4);
    float* hbuf    = (float*)(ws + off); off = al256(off + (size_t)MROWS * CMID * 4);
    float* tbuf    = (float*)(ws + off); off = al256(off + (size_t)MROWS * CMID * 4);
    float* vbuf    = (float*)(ws + off); size_t vbytes = (size_t)NB * HP * WP * CMID * 4;
    off = al256(off + vbytes);
    float* ombuf   = (float*)(ws + off); off = al256(off + (size_t)MROWS * 128 * 4);
    float* sbuf    = (float*)(ws + off); off = al256(off + (size_t)MROWS * CMID * 4);

    // 1) weight repack (transposed) + padded activation layout transform
    hipMemsetAsync(xh, 0, xhbytes, stream);    // zero spatial border
    hipMemsetAsync(vbuf, 0, vbytes, stream);   // zero padding border for value
    prep_weights<<<((128 * 2304 + 128 * 128 * 2 + 256 * 128 + 128) + 255) / 256, 256, 0, stream>>>(
        w_cv1, w_inp, w_off, w_msk, w_out, b_off, b_msk, Bt1, Bt2, Bt3, Bt4, bias_om);
    x_to_nhwc<<<(MROWS * C1C) / 256, 256, 0, stream>>>(x, xh);

    // 2) cv1: implicit-GEMM conv + BN1 + SiLU -> h
    gemm_wmma<0><<<dim3(1, MROWS / 64), 256, 0, stream>>>(
        hbuf /*unused*/, xh, Bt1, 2304, nullptr,
        bn1_g, bn1_b, bn1_m, bn1_v, nullptr, hbuf);

    // 3) depthwise + LN + GELU -> t
    dw_ln_gelu<<<MROWS / 8, 256, 0, stream>>>(hbuf, w_dw, b_dw, ln_g, ln_b, tbuf);

    // 4) offset/mask logits GEMM (t @ [w_off|w_msk])
    gemm_wmma<2><<<dim3(1, MROWS / 64), 256, 0, stream>>>(
        tbuf, xh, Bt3, 128, bias_om,
        nullptr, nullptr, nullptr, nullptr, nullptr, ombuf);

    // 5) input_proj GEMM -> padded value
    gemm_wmma<1><<<dim3(1, MROWS / 64), 256, 0, stream>>>(
        hbuf, xh, Bt2, 128, b_inp,
        nullptr, nullptr, nullptr, nullptr, nullptr, vbuf);

    // 6) deformable sampling
    dcn_sample<<<(MROWS * GRP) / 8, 256, 0, stream>>>(ombuf, vbuf, sbuf);

    // 7) output_proj GEMM + BN2 + SiLU + residual -> d_out (NCHW f32)
    gemm_wmma<3><<<dim3(2, MROWS / 64), 256, 0, stream>>>(
        sbuf, xh, Bt4, 128, b_out,
        bn2_g, bn2_b, bn2_m, bn2_v, x, out);
}
